// RNNModel_32641751450084
// MI455X (gfx1250) — matrix operand order repaired and checked by reading, not compiled, not run on hardware
//
#include <hip/hip_runtime.h>

// Problem constants (from the reference): B=8192, T=80, D=32, H=64, L=3
#define B_SZ 8192
#define T_SZ 80
#define D_SZ 32
#define H_SZ 64

typedef __attribute__((ext_vector_type(16))) __bf16        v16bf;
typedef __attribute__((ext_vector_type(8)))  float         v8f;
typedef __attribute__((ext_vector_type(4)))  unsigned int  u32x4;
typedef __attribute__((ext_vector_type(4)))  float         f32x4;

struct FragBits { u32x4 lo, hi; };
static_assert(sizeof(FragBits) == sizeof(v16bf), "fragment size mismatch");

// ---- bf16 helpers ----
__device__ __forceinline__ float bf2f(unsigned short b) {
  return __builtin_bit_cast(float, (unsigned)b << 16);
}

// Pack two f32 into bf16x2 using CDNA5 VOP3P V_FMA_MIXLO_BF16 / V_FMA_MIXHI_BF16
// (ISA §15.10 opcodes 62/63): D[15:0]/D[31:16] = cvt_bf16(fma(src0, 1.0, 0)).
// Confirmed to assemble on gfx1250 (round 4). Two VALU ops per pair, result
// already packed into register halves.
__device__ __forceinline__ unsigned pack2(float lo, float hi) {
  unsigned pk;
  asm("v_fma_mixlo_bf16 %0, %1, 1.0, 0" : "=v"(pk) : "v"(lo));
  asm("v_fma_mixhi_bf16 %0, %1, 1.0, 0" : "+v"(pk) : "v"(hi));
  return pk;
}

// Branch-free tanh: gfx1250 hardware v_tanh_f32 (confirmed emitted), with
// overflow-safe exp2/rcp fallback.
__device__ __forceinline__ float fast_tanh(float x) {
#if __has_builtin(__builtin_amdgcn_tanhf)
  return __builtin_amdgcn_tanhf(x);
#else
  float ax = __builtin_fabsf(x);
#if __has_builtin(__builtin_amdgcn_exp2f)
  float e = __builtin_amdgcn_exp2f(ax * 2.8853900817779268f);
#else
  float e = __builtin_exp2f(ax * 2.8853900817779268f);
#endif
#if __has_builtin(__builtin_amdgcn_rcpf)
  float r = 1.0f - 2.0f * __builtin_amdgcn_rcpf(e + 1.0f);
#else
  float r = 1.0f - 2.0f / (e + 1.0f);
#endif
  return __builtin_copysignf(r, x);
#endif
}

// A-fragment (16x32 bf16) from LDS. ISA A layout: e -> K = 16*(e/8) + 8*half + e%8.
// p must already include: row(l15)*64 + kc*32 + half*8 (ushort units)
// => two contiguous 16B segments at +0 and +16.
__device__ __forceinline__ v16bf frag_a_lds(const unsigned short* p) {
  FragBits fb;
  fb.lo = *reinterpret_cast<const u32x4*>(p);
  fb.hi = *reinterpret_cast<const u32x4*>(p + 16);
  return __builtin_bit_cast(v16bf, fb);
}

// B-fragment (32x16 bf16) from LDS bf16 weights, row-major W[n][k].
// ISA B layout: lane = N (mod 16), element e -> K = 16*(e/8) + 8*half + e%8
// => two contiguous 16B segments at +0 and +16, p = W + n*Kdim + kc*32 + half*8.
__device__ __forceinline__ v16bf frag_b_lds(const unsigned short* p) {
  FragBits fb;
  fb.lo = *reinterpret_cast<const u32x4*>(p);
  fb.hi = *reinterpret_cast<const u32x4*>(p + 16);
  return __builtin_bit_cast(v16bf, fb);
}

// A-fragment (16x32) from global f32 x; p = row_base + half*8 (float units);
// segments at +0..7 and +16..23 per the A layout.
__device__ __forceinline__ v16bf frag_a_global_f32(const float* p) {
  f32x4 a = *reinterpret_cast<const f32x4*>(p);
  f32x4 b = *reinterpret_cast<const f32x4*>(p + 4);
  f32x4 c = *reinterpret_cast<const f32x4*>(p + 16);
  f32x4 d = *reinterpret_cast<const f32x4*>(p + 20);
  FragBits fb;
  u32x4 lo = {pack2(a.x, a.y), pack2(a.z, a.w), pack2(b.x, b.y), pack2(b.z, b.w)};
  u32x4 hi = {pack2(c.x, c.y), pack2(c.z, c.w), pack2(d.x, d.y), pack2(d.z, d.w)};
  fb.lo = lo; fb.hi = hi;
  return __builtin_bit_cast(v16bf, fb);
}

// LDS layout: bf16 weights (44 KB, shared by both waves) + per-wave h state.
struct SMem {
  alignas(16) unsigned int Wih0[H_SZ * D_SZ / 2];   // 4 KB  (bf16 pairs)
  alignas(16) unsigned int Whh0[H_SZ * H_SZ / 2];   // 8 KB
  alignas(16) unsigned int Wih1[H_SZ * H_SZ / 2];   // 8 KB
  alignas(16) unsigned int Whh1[H_SZ * H_SZ / 2];   // 8 KB
  alignas(16) unsigned int Wih2[H_SZ * H_SZ / 2];   // 8 KB
  alignas(16) unsigned int Whh2[H_SZ * H_SZ / 2];   // 8 KB
  alignas(16) unsigned short h[2][3][16 * H_SZ];    // 12 KB (per-wave)
};

// Stage f32 weights -> packed bf16 pairs in LDS (one v_fma_mixlo+hi per pair).
__device__ __forceinline__ void stage_bf16(unsigned int* dst, const float* src,
                                           int npairs, int tid, int nt) {
  for (int i = tid; i < npairs; i += nt)
    dst[i] = pack2(src[2 * i], src[2 * i + 1]);
}

// One fused RNN layer step for a 16-row batch tile.
// Weights in LDS (bf16); hbuf (LDS) holds h and provides the
// C/D-layout -> A-layout transpose between timesteps.
template <int NIN>
__device__ __forceinline__ void rnn_layer_step(
    const unsigned short* __restrict__ Wih, int din,
    const unsigned short* __restrict__ Whh,
    const float* bs,           // [4] per-chunk bias value for this lane
    const v16bf* a_in,         // [NIN]
    unsigned short* __restrict__ hbuf,
    unsigned l15, unsigned half)
{
  const unsigned hb   = half * 8;
  const unsigned hoff = half * 16;
  // h_{t-1} fragments (read before overwriting)
  v16bf ah0 = frag_a_lds(hbuf + l15 * H_SZ + 0  + hb);
  v16bf ah1 = frag_a_lds(hbuf + l15 * H_SZ + 32 + hb);
#pragma unroll
  for (int n0 = 0; n0 < 4; ++n0) {
    const unsigned n = (unsigned)n0 * 16 + l15;
    const float bn = bs[n0];
    v8f c = {bn, bn, bn, bn, bn, bn, bn, bn};
#pragma unroll
    for (int kc = 0; kc < NIN; ++kc)
      c = __builtin_amdgcn_wmma_f32_16x16x32_bf16(
          false, a_in[kc], false,
          frag_b_lds(Wih + n * din + kc * 32 + hb), (short)0, c, false, false);
    c = __builtin_amdgcn_wmma_f32_16x16x32_bf16(
        false, ah0, false, frag_b_lds(Whh + n * H_SZ + 0  + hb),
        (short)0, c, false, false);
    c = __builtin_amdgcn_wmma_f32_16x16x32_bf16(
        false, ah1, false, frag_b_lds(Whh + n * H_SZ + 32 + hb),
        (short)0, c, false, false);
    // v_tanh_f32 + v_fma_mixlo/hi_bf16 packed convert; store low half as b16
    // and high half via (pk>>16) (ds_store_b16_d16_hi pattern).
#pragma unroll
    for (int r = 0; r < 8; r += 2) {
      unsigned pk = pack2(fast_tanh(c[r]), fast_tanh(c[r + 1]));
      hbuf[(r + 0 + hb) * H_SZ + n] = (unsigned short)pk;
      hbuf[(r + 1 + hb) * H_SZ + n] = (unsigned short)(pk >> 16);
    }
  }
}

__global__ __launch_bounds__(64, 1) void rnn3_wmma_kernel(
    const float* __restrict__ x,
    const float* __restrict__ gWih0, const float* __restrict__ gWhh0,
    const float* __restrict__ gbih0, const float* __restrict__ gbhh0,
    const float* __restrict__ gWih1, const float* __restrict__ gWhh1,
    const float* __restrict__ gbih1, const float* __restrict__ gbhh1,
    const float* __restrict__ gWih2, const float* __restrict__ gWhh2,
    const float* __restrict__ gbih2, const float* __restrict__ gbhh2,
    const float* __restrict__ gWfc, const float* __restrict__ gbfc,
    float* __restrict__ out)
{
  __shared__ SMem sm;
  const int tid = (int)threadIdx.x;
  const int nt  = (int)blockDim.x;   // 64 = 2 waves

  // Stage all weights as packed bf16 into LDS once (shared by both waves).
  stage_bf16(sm.Wih0, gWih0, H_SZ * D_SZ / 2, tid, nt);
  stage_bf16(sm.Whh0, gWhh0, H_SZ * H_SZ / 2, tid, nt);
  stage_bf16(sm.Wih1, gWih1, H_SZ * H_SZ / 2, tid, nt);
  stage_bf16(sm.Whh1, gWhh1, H_SZ * H_SZ / 2, tid, nt);
  stage_bf16(sm.Wih2, gWih2, H_SZ * H_SZ / 2, tid, nt);
  stage_bf16(sm.Whh2, gWhh2, H_SZ * H_SZ / 2, tid, nt);

  const unsigned wave = (unsigned)tid >> 5;
  const unsigned lane = (unsigned)tid & 31u;
  const unsigned l15  = lane & 15u;
  const unsigned half = (lane >> 4) & 1u;

  unsigned short* h0 = sm.h[wave][0];
  unsigned short* h1 = sm.h[wave][1];
  unsigned short* h2 = sm.h[wave][2];
  // h(-1) = 0 for this wave's three layers
  {
    u32x4 z = {0u, 0u, 0u, 0u};
    u32x4* p = reinterpret_cast<u32x4*>(&sm.h[wave][0][0]);
    for (int i = (int)lane; i < (3 * 16 * H_SZ) / 8; i += 32) p[i] = z;
  }

  // Per-lane biases in registers (b_ih + b_hh pre-summed).
  float bs0[4], bs1[4], bs2[4];
#pragma unroll
  for (int n0 = 0; n0 < 4; ++n0) {
    const unsigned n = (unsigned)n0 * 16 + l15;
    bs0[n0] = gbih0[n] + gbhh0[n];
    bs1[n0] = gbih1[n] + gbhh1[n];
    bs2[n0] = gbih2[n] + gbhh2[n];
  }

  __syncthreads();   // weights staged by both waves

  const unsigned short* W0 = (const unsigned short*)sm.Wih0;
  const unsigned short* U0 = (const unsigned short*)sm.Whh0;
  const unsigned short* W1 = (const unsigned short*)sm.Wih1;
  const unsigned short* U1 = (const unsigned short*)sm.Whh1;
  const unsigned short* W2 = (const unsigned short*)sm.Wih2;
  const unsigned short* U2 = (const unsigned short*)sm.Whh2;

  const int m0 = (int)blockIdx.x * 32 + (int)wave * 16;  // batch tile base
  const float* xrow = x + ((size_t)(m0 + (int)l15) * T_SZ) * D_SZ + half * 8;

  for (int t = 0; t < T_SZ; ++t) {
    // Prefetch next timestep's x row (global_prefetch_b8); speculative
    // past-the-end prefetch is dropped harmlessly.
    __builtin_prefetch(xrow + (t + 1) * D_SZ, 0, 1);

    // layer 0: input from global x (K=32 -> one A fragment)
    v16bf ax[1] = { frag_a_global_f32(xrow + t * D_SZ) };
    rnn_layer_step<1>(W0, D_SZ, U0, bs0, ax, h0, l15, half);

    // layer 1: input = h0(t) from LDS (K=64 -> two fragments)
    v16bf a1[2] = { frag_a_lds(h0 + l15 * H_SZ + 0  + half * 8),
                    frag_a_lds(h0 + l15 * H_SZ + 32 + half * 8) };
    rnn_layer_step<2>(W1, H_SZ, U1, bs1, a1, h1, l15, half);

    // layer 2: input = h1(t)
    v16bf a2[2] = { frag_a_lds(h1 + l15 * H_SZ + 0  + half * 8),
                    frag_a_lds(h1 + l15 * H_SZ + 32 + half * 8) };
    rnn_layer_step<2>(W2, H_SZ, U2, bs2, a2, h2, l15, half);
  }

  // Final FC: out[b] = h2[b,:] . Wfc + bfc  (lanes 0..15, one batch row each)
  if (lane < 16) {
    const unsigned short* row = h2 + lane * H_SZ;
    float acc = gbfc[0];
#pragma unroll
    for (int k = 0; k < H_SZ; ++k) acc += bf2f(row[k]) * gWfc[k];
    out[m0 + (int)lane] = acc;
  }
}

extern "C" void kernel_launch(void* const* d_in, const int* in_sizes, int n_in,
                              void* d_out, int out_size, void* d_ws, size_t ws_size,
                              hipStream_t stream) {
  (void)in_sizes; (void)n_in; (void)out_size; (void)d_ws; (void)ws_size;
  const float* x    = (const float*)d_in[0];
  const float* Wih0 = (const float*)d_in[1];
  const float* Whh0 = (const float*)d_in[2];
  const float* bih0 = (const float*)d_in[3];
  const float* bhh0 = (const float*)d_in[4];
  const float* Wih1 = (const float*)d_in[5];
  const float* Whh1 = (const float*)d_in[6];
  const float* bih1 = (const float*)d_in[7];
  const float* bhh1 = (const float*)d_in[8];
  const float* Wih2 = (const float*)d_in[9];
  const float* Whh2 = (const float*)d_in[10];
  const float* bih2 = (const float*)d_in[11];
  const float* bhh2 = (const float*)d_in[12];
  const float* Wfc  = (const float*)d_in[13];
  const float* bfc  = (const float*)d_in[14];

  dim3 grid(B_SZ / 32);   // 256 blocks x 2 waves = 512 independent chains
  dim3 block(64);
  rnn3_wmma_kernel<<<grid, block, 0, stream>>>(
      x, Wih0, Whh0, bih0, bhh0, Wih1, Whh1, bih1, bhh1,
      Wih2, Whh2, bih2, bhh2, Wfc, bfc, (float*)d_out);
}
